// MultiHeadAttention_26946624815802
// MI455X (gfx1250) — matrix operand order repaired and checked by reading, not compiled, not run on hardware
//
#include <hip/hip_runtime.h>

#define DMODEL 1024
#define SEQ    2048
#define NH     16
#define HD     64
#define LOG2E  1.4426950408889634f

typedef __attribute__((ext_vector_type(16))) __bf16        v16bf;
typedef __attribute__((ext_vector_type(8)))  float         v8f;
typedef __attribute__((ext_vector_type(4)))  unsigned int  u32x4;
typedef __attribute__((ext_vector_type(4)))  float         f32x4;

union V16 { v16bf v; u32x4 u[2]; };

// ---- DPP16 row-rotate cross-lane reduction (wave32: rows of 16 lanes) ------
template <int N>  // rotate amount: ctrl = ROW_ROR:N = 0x120 | N
__device__ __forceinline__ float row_ror(float v) {
    int i = __builtin_bit_cast(int, v);
    int r = __builtin_amdgcn_update_dpp(i, i, 0x120 | N, 0xf, 0xf, false);
    return __builtin_bit_cast(float, r);
}
__device__ __forceinline__ float row_max16(float v) {
    v = fmaxf(v, row_ror<1>(v));
    v = fmaxf(v, row_ror<2>(v));
    v = fmaxf(v, row_ror<4>(v));
    v = fmaxf(v, row_ror<8>(v));
    return v;
}
__device__ __forceinline__ float row_sum16(float v) {
    v += row_ror<1>(v);
    v += row_ror<2>(v);
    v += row_ror<4>(v);
    v += row_ror<8>(v);
    return v;
}
__device__ __forceinline__ float fast_expf(float x) {   // e^x, x <= 0 here
    return __builtin_amdgcn_exp2f(x * LOG2E);
}
// wave-uniform 64-bit pointer forced into SGPRs
__device__ __forceinline__ unsigned long long scalar_ptr(const void* p) {
    unsigned lo = __builtin_amdgcn_readfirstlane((unsigned)(uintptr_t)p);
    unsigned hi = __builtin_amdgcn_readfirstlane((unsigned)((uintptr_t)p >> 32));
    return ((unsigned long long)hi << 32) | lo;
}

// ---------------------------------------------------------------------------
// Weight prep: W[k][n] f32  ->  Wt[n][k] bf16  (B-fragment friendly layout)
// ---------------------------------------------------------------------------
__global__ __launch_bounds__(256) void transpose_cast_bf16(
        const float* __restrict__ in, __bf16* __restrict__ out) {
    size_t id = (size_t)blockIdx.x * blockDim.x + threadIdx.x;
    int n = (int)(id / DMODEL);
    int k = (int)(id % DMODEL);
    out[(size_t)n * DMODEL + k] = (__bf16)in[(size_t)k * DMODEL + n];
}

// ---------------------------------------------------------------------------
// WMMA GEMM: C[M,1024] = A[M,1024] * Wt^T + bias.
// One wave computes a 32x64 tile: 2 A-fragments x 4 B-fragments = 8 WMMAs
// per 32-wide k-step (B fragments reused across both row tiles).
// MODE 0: store bf16 to (b,h,seq,64)          (Q and K projections)
// MODE 1: store bf16 to (b,h,64,seq)          (V projection, transposed)
// MODE 2: store f32 = acc + bias + residual   (output projection)
// ---------------------------------------------------------------------------
template <int MODE, bool AF32>
__global__ __launch_bounds__(128, 1) void gemm_wmma(
                          const void* __restrict__ Ap,
                          const __bf16* __restrict__ Bt,
                          const float* __restrict__ bias,
                          void* __restrict__ Cp,
                          const float* __restrict__ resid,
                          int M, int Kdim, int Nstrips) {
    const int lane = threadIdx.x & 31;
    const int wv   = threadIdx.x >> 5;
    const int wid  = blockIdx.x * (blockDim.x >> 5) + wv;
    const int mt   = wid / Nstrips;
    const int ns   = wid % Nstrips;
    const int m0   = mt * 32;
    const int n0   = ns * 64;
    if (m0 >= M) return;                      // wave-uniform guard (EXEC all-1)

    const int nl   = lane & 15;
    const int half = lane >> 4;

    v8f acc[2][4] = {};

    for (int kk = 0; kk < Kdim; kk += 32) {
        // ---- two A fragments: rows m0+nl and m0+16+nl
        V16 af[2];
#pragma unroll
        for (int mi = 0; mi < 2; ++mi) {
            if (AF32) {
                const float* p0 = (const float*)Ap
                    + (size_t)(m0 + mi * 16 + nl) * Kdim + kk + 8 * half;
                f32x4 a0 = *(const f32x4*)(p0);
                f32x4 a1 = *(const f32x4*)(p0 + 4);
                f32x4 a2 = *(const f32x4*)(p0 + 16);
                f32x4 a3 = *(const f32x4*)(p0 + 20);
#pragma unroll
                for (int e = 0; e < 4; ++e) {
                    af[mi].v[e]      = (__bf16)a0[e];
                    af[mi].v[4 + e]  = (__bf16)a1[e];
                    af[mi].v[8 + e]  = (__bf16)a2[e];
                    af[mi].v[12 + e] = (__bf16)a3[e];
                }
            } else {
                const __bf16* p0 = (const __bf16*)Ap
                    + (size_t)(m0 + mi * 16 + nl) * Kdim + kk + 8 * half;
                af[mi].u[0] = *(const u32x4*)(p0);
                af[mi].u[1] = *(const u32x4*)(p0 + 16);
            }
        }
        // ---- 4 B fragments, each reused for both row tiles
#pragma unroll
        for (int t = 0; t < 4; ++t) {
            const __bf16* bp = Bt + (size_t)(n0 + t * 16 + nl) * Kdim + kk + 8 * half;
            V16 bf;
            bf.u[0] = *(const u32x4*)(bp);
            bf.u[1] = *(const u32x4*)(bp + 16);
            acc[0][t] = __builtin_amdgcn_wmma_f32_16x16x32_bf16(
                false, af[0].v, false, bf.v, (short)0, acc[0][t], false, false);
            acc[1][t] = __builtin_amdgcn_wmma_f32_16x16x32_bf16(
                false, af[1].v, false, bf.v, (short)0, acc[1][t], false, false);
        }
    }

    // ---- epilogue
#pragma unroll
    for (int t = 0; t < 4; ++t) {
        const int gn = n0 + t * 16 + nl;
        const float bb = bias[gn];
#pragma unroll
        for (int mi = 0; mi < 2; ++mi)
#pragma unroll
            for (int r = 0; r < 8; ++r) {
                const int gm = m0 + mi * 16 + r + 8 * half;
                float val = acc[mi][t][r] + bb;
                if (MODE == 0) {
                    int b = gm / SEQ, s = gm % SEQ, h = gn / HD, d = gn % HD;
                    ((__bf16*)Cp)[(((size_t)b * NH + h) * SEQ + s) * HD + d] = (__bf16)val;
                } else if (MODE == 1) {
                    int b = gm / SEQ, s = gm % SEQ, h = gn / HD, d = gn % HD;
                    ((__bf16*)Cp)[(((size_t)b * NH + h) * HD + d) * SEQ + s] = (__bf16)val;
                } else {
                    size_t idx = (size_t)gm * DMODEL + gn;
                    ((float*)Cp)[idx] = val + resid[idx];
                }
            }
    }
}

// ---------------------------------------------------------------------------
// Async staging of one V tile (64 dv rows x 32 keys, bf16 = 4 KB) to LDS.
// 8 x global_load_async_to_lds_b128 in a rolled loop (low VGPR pressure):
// lane copies 16 B; row = lane/4 (+8 per step), chunk = lane&3. The LDS chunk
// index is XOR-swizzled by (row>>2)&3 so both the async writes and the later
// ds_load_b128 fragment reads sweep all 64 banks conflict-free.
// ---------------------------------------------------------------------------
__device__ __forceinline__ void stage_v_async(unsigned long long sa,
                                              unsigned jbytes,
                                              unsigned ldsbyte, int lane) {
    const unsigned r0 = (unsigned)(lane >> 2);
    const unsigned c  = (unsigned)(lane & 3);
#pragma unroll 1
    for (unsigned r = r0; r < 64u; r += 8u) {
        unsigned phys = c ^ ((r >> 2) & 3u);
        unsigned lds  = ldsbyte + r * 64u + phys * 16u;
        unsigned voff = jbytes + r * (SEQ * 2u) + c * 16u;
        asm volatile("global_load_async_to_lds_b128 %0, %1, %2"
                     :: "v"(lds), "v"(voff), "s"(sa) : "memory");
    }
}

// ---------------------------------------------------------------------------
// Flash attention: one wave = one (b, h, 16-query tile).
// Per 32-key chunk: 4 WMMAs for S, DPP-rotate online softmax (O-rescale
// fused per row), P transposed C->A layout through padded LDS, O += P*V via
// 4 WMMAs reading the async-staged, double-buffered V tile from LDS.
// ---------------------------------------------------------------------------
__global__ __launch_bounds__(128, 1) void attn_flash(
                           const __bf16* __restrict__ qp,
                           const __bf16* __restrict__ kp,
                           const __bf16* __restrict__ vT,
                           const int* __restrict__ mask,
                           __bf16* __restrict__ aout) {
    const int lane = threadIdx.x & 31;
    const int wv   = threadIdx.x >> 5;
    const int wid  = blockIdx.x * (blockDim.x >> 5) + wv;
    const int qt   = wid & (SEQ / 16 - 1);
    const int bh   = wid / (SEQ / 16);
    const int b    = bh / NH;
    const int h    = bh % NH;
    const int q0   = qt * 16;
    const int nl   = lane & 15;
    const int half = lane >> 4;

    __shared__ __bf16 plds[4][16][34];                       // P transpose pad
    __shared__ __align__(16) __bf16 vlds[4][2][64 * 32];     // V double buffer

    const unsigned long long vsa = scalar_ptr(vT + (size_t)bh * HD * SEQ);
    const unsigned ldsb0 = (unsigned)(uintptr_t)&vlds[wv][0][0];
    const unsigned ldsb1 = (unsigned)(uintptr_t)&vlds[wv][1][0];

    // Q fragments for full dk=64 (two 32-k chunks), resident whole kernel
    V16 qf[2];
#pragma unroll
    for (int c = 0; c < 2; ++c) {
        const __bf16* p0 = qp + ((size_t)bh * SEQ + q0 + nl) * HD + c * 32 + 8 * half;
        qf[c].u[0] = *(const u32x4*)(p0);
        qf[c].u[1] = *(const u32x4*)(p0 + 16);
    }

    v8f O[4] = {};
    float mrow[8], lrow[8];
#pragma unroll
    for (int r = 0; r < 8; ++r) { mrow[r] = -1e30f; lrow[r] = 0.0f; }

    // prologue: stage first V tile into buffer 0
    stage_v_async(vsa, 0u, ldsb0, lane);

    for (int j = 0; j < SEQ; j += 32) {
        const int buf = (j >> 5) & 1;
        const unsigned jn = (unsigned)((j + 32) & (SEQ - 1));   // wrapped
        stage_v_async(vsa, jn * 2u, buf ? ldsb0 : ldsb1, lane);
        __builtin_prefetch(kp + ((size_t)bh * SEQ + jn) * HD, 0, 3);

        // ---- S = Q K^T (two 16x16 tiles over keys j..j+31)
        v8f S0 = {}, S1 = {};
#pragma unroll
        for (int c = 0; c < 2; ++c) {
            const __bf16* k0 = kp + ((size_t)bh * SEQ + j + nl) * HD + c * 32 + 8 * half;
            const __bf16* k1 = kp + ((size_t)bh * SEQ + j + 16 + nl) * HD + c * 32 + 8 * half;
            V16 kf0, kf1;
            kf0.u[0] = *(const u32x4*)(k0);  kf0.u[1] = *(const u32x4*)(k0 + 16);
            kf1.u[0] = *(const u32x4*)(k1);  kf1.u[1] = *(const u32x4*)(k1 + 16);
            S0 = __builtin_amdgcn_wmma_f32_16x16x32_bf16(false, qf[c].v, false, kf0.v,
                                                         (short)0, S0, false, false);
            S1 = __builtin_amdgcn_wmma_f32_16x16x32_bf16(false, qf[c].v, false, kf1.v,
                                                         (short)0, S1, false, false);
        }
        const int mv0 = mask[(size_t)b * SEQ + j + nl];
        const int mv1 = mask[(size_t)b * SEQ + j + 16 + nl];

        // ---- online softmax per row r (+8*half); O-rescale fused, pure VALU
#pragma unroll
        for (int r = 0; r < 8; ++r) {
            float t0 = mv0 ? S0[r] * 0.125f : -1e30f;
            float t1 = mv1 ? S1[r] * 0.125f : -1e30f;
            float cm   = row_max16(fmaxf(t0, t1));
            float mnew = fmaxf(mrow[r], cm);
            float al   = fast_expf(mrow[r] - mnew);
            float p0   = fast_expf(t0 - mnew);
            float p1   = fast_expf(t1 - mnew);
            lrow[r] = lrow[r] * al + row_sum16(p0 + p1);
            mrow[r] = mnew;
            plds[wv][r + 8 * half][nl]      = (__bf16)p0;
            plds[wv][r + 8 * half][16 + nl] = (__bf16)p1;
            O[0][r] = O[0][r] * al;
            O[1][r] = O[1][r] * al;
            O[2][r] = O[2][r] * al;
            O[3][r] = O[3][r] * al;
        }

        // ---- reload P in A-fragment layout (same-wave LDS is in-order)
        V16 pf;
#pragma unroll
        for (int i = 0; i < 8; ++i) {
            int k = 16 * half + 2 * i;
            pf.v[2 * i]     = plds[wv][nl][k];
            pf.v[2 * i + 1] = plds[wv][nl][k + 1];
        }

        // ---- oldest 8 async loads = this iteration's V tile (in-order)
        asm volatile("s_wait_asynccnt 0x8" ::: "memory");

        // ---- batch all 4 V fragments (8x ds_load_b128), then 4 WMMAs
        V16 vf[4];
#pragma unroll
        for (int t = 0; t < 4; ++t) {
            const int dv = t * 16 + nl;
            const int sw = (dv >> 2) & 3;
            const __bf16* rowp = &vlds[wv][buf][dv * 32];
            vf[t].u[0] = *(const u32x4*)(rowp + ((2 * half) ^ sw) * 8);
            vf[t].u[1] = *(const u32x4*)(rowp + ((2 * half + 1) ^ sw) * 8);
        }
#pragma unroll
        for (int t = 0; t < 4; ++t)
            O[t] = __builtin_amdgcn_wmma_f32_16x16x32_bf16(false, pf.v, false, vf[t].v,
                                                           (short)0, O[t], false, false);
    }

    // ---- normalize by row sums, store (b, q, h*64+dv) bf16
#pragma unroll
    for (int r = 0; r < 8; ++r) {
        float inv = lrow[r] > 0.0f ? __builtin_amdgcn_rcpf(lrow[r]) : 0.0f;
        size_t rowbase = ((size_t)b * SEQ + q0 + r + 8 * half) * DMODEL + h * HD + nl;
#pragma unroll
        for (int t = 0; t < 4; ++t)
            aout[rowbase + t * 16] = (__bf16)(O[t][r] * inv);
    }
}

// ---------------------------------------------------------------------------
// LayerNorm over d_model=1024: one 256-thread block per row.
// DPP 16-lane row reductions + 16 partials through LDS (no bpermute).
// ---------------------------------------------------------------------------
__global__ __launch_bounds__(256) void layernorm_k(
                            const float* __restrict__ x,
                            const float* __restrict__ gamma,
                            const float* __restrict__ beta,
                            float* __restrict__ out) {
    const int row = blockIdx.x;
    const int t   = threadIdx.x;
    const float* xr = x + (size_t)row * DMODEL;
    float s = 0.0f, ss = 0.0f;
    for (int i = t; i < DMODEL; i += 256) {
        float v = xr[i];
        s += v; ss += v * v;
    }
    s  = row_sum16(s);
    ss = row_sum16(ss);
    __shared__ float sh[2][16];
    if ((t & 15) == 0) { sh[0][t >> 4] = s; sh[1][t >> 4] = ss; }
    __syncthreads();
    float ts = 0.0f, tss = 0.0f;
#pragma unroll
    for (int w = 0; w < 16; ++w) { ts += sh[0][w]; tss += sh[1][w]; }
    float mean = ts * (1.0f / DMODEL);
    float var  = tss * (1.0f / DMODEL) - mean * mean;
    float invs = rsqrtf(var + 1e-5f);
    for (int i = t; i < DMODEL; i += 256)
        out[(size_t)row * DMODEL + i] = gamma[i] * (xr[i] - mean) * invs + beta[i];
}

// ---------------------------------------------------------------------------
extern "C" void kernel_launch(void* const* d_in, const int* in_sizes, int n_in,
                              void* d_out, int out_size, void* d_ws, size_t ws_size,
                              hipStream_t stream) {
    const float* queries = (const float*)d_in[0];
    const float* keys    = (const float*)d_in[1];
    const float* values  = (const float*)d_in[2];
    const int*   amask   = (const int*)  d_in[3];
    const float* Wq = (const float*)d_in[4];  const float* bq = (const float*)d_in[5];
    const float* Wk = (const float*)d_in[6];  const float* bk = (const float*)d_in[7];
    const float* Wv = (const float*)d_in[8];  const float* bv = (const float*)d_in[9];
    const float* Wo = (const float*)d_in[10]; const float* bo = (const float*)d_in[11];
    const float* gamma = (const float*)d_in[12];
    const float* beta  = (const float*)d_in[13];

    const int B = in_sizes[0] / (SEQ * DMODEL);   // 4
    const int M = B * SEQ;                        // 8192

    // workspace carve-up
    char* w = (char*)d_ws;
    const size_t WSZ = (size_t)DMODEL * DMODEL * sizeof(__bf16);   // 2 MB
    __bf16* WqT  = (__bf16*)(w);            w += WSZ;
    __bf16* WkT  = (__bf16*)(w);            w += WSZ;
    __bf16* WvT  = (__bf16*)(w);            w += WSZ;
    __bf16* WoT  = (__bf16*)(w);            w += WSZ;
    const size_t PSZ = (size_t)M * DMODEL * sizeof(__bf16);        // 16 MB
    __bf16* qp   = (__bf16*)(w);            w += PSZ;
    __bf16* kpj  = (__bf16*)(w);            w += PSZ;
    __bf16* vTp  = (__bf16*)(w);            w += PSZ;
    __bf16* aout = (__bf16*)(w);            w += PSZ;
    float*  xbuf = (float*)(w);             // 32 MB

    // 1) weight prep (transpose + cast to bf16)
    const int tblocks = (DMODEL * DMODEL) / 256;
    transpose_cast_bf16<<<tblocks, 256, 0, stream>>>(Wq, WqT);
    transpose_cast_bf16<<<tblocks, 256, 0, stream>>>(Wk, WkT);
    transpose_cast_bf16<<<tblocks, 256, 0, stream>>>(Wv, WvT);
    transpose_cast_bf16<<<tblocks, 256, 0, stream>>>(Wo, WoT);

    // 2) input projections (WMMA bf16, f32 accumulate), 32x64 tiles per wave
    const int strips  = DMODEL / 64;                  // 16
    const int waves   = (M / 32) * strips;            // 4096
    const int gblocks = waves / 4;                    // 4 waves per 128-thr block
    gemm_wmma<0, true><<<gblocks, 128, 0, stream>>>(queries, WqT, bq, qp,  nullptr, M, DMODEL, strips);
    gemm_wmma<0, true><<<gblocks, 128, 0, stream>>>(keys,    WkT, bk, kpj, nullptr, M, DMODEL, strips);
    gemm_wmma<1, true><<<gblocks, 128, 0, stream>>>(values,  WvT, bv, vTp, nullptr, M, DMODEL, strips);

    // 3) flash attention (one wave per 16-query tile per head)
    const int awaves  = B * NH * (SEQ / 16);          // 8192
    attn_flash<<<awaves / 4, 128, 0, stream>>>(qp, kpj, vTp, amask, aout);

    // 4) output projection + bias + residual (f32 out)
    gemm_wmma<2, false><<<gblocks, 128, 0, stream>>>(aout, WoT, bo, xbuf, queries, M, DMODEL, strips);

    // 5) LayerNorm -> d_out
    layernorm_k<<<M, 256, 0, stream>>>(xbuf, gamma, beta, (float*)d_out);
}
